// ViTMAESelfAttention_60309930770575
// MI455X (gfx1250) — matrix operand order repaired and checked by reading, not compiled, not run on hardware
//
#include <hip/hip_runtime.h>
#include <hip/hip_bf16.h>
#include <math.h>

// ---------------------------------------------------------------------------
// ViT-MAE self-attention for MI455X (gfx1250, wave32, WMMA, async-LDS).
//   B=8, S=1024, HID=1024, HEADS=16, HD=64
// Pipeline:
//   1) qkv_proj : Q/K/V = X*W + b, f32 in -> f16 out (workspace), WMMA f16
//                 128x256 block tile, 64x64 wave tile (16 WMMA / k-step / wave)
//   2) attention: flash-attention, 64-key blocks, async K/Q staging to LDS
// ---------------------------------------------------------------------------

typedef __attribute__((ext_vector_type(16))) _Float16 v16h;
typedef __attribute__((ext_vector_type(8)))  _Float16 v8h;
typedef __attribute__((ext_vector_type(4)))  _Float16 v4h;
typedef __attribute__((ext_vector_type(8)))  float    v8f;

union AFrag { v16h v; v8h h[2]; };

#define HID   1024
#define SEQ   1024
#define BATCH 8
#define HEADS 16
#define HD    64
#define MROWS (BATCH * SEQ)   // 8192

static __device__ inline v8f wmma_f16(v16h a, v16h b, v8f c) {
    return __builtin_amdgcn_wmma_f32_16x16x32_f16(false, a, false, b,
                                                  (short)0, c, false, false);
}

// Low 32 bits of a generic pointer to LDS == LDS byte offset (ISA 10.2).
static __device__ inline unsigned lds_off(const void* p) {
    return (unsigned)(unsigned long long)p;
}

// ---------------------------------------------------------------------------
// Kernel 1: QKV projection.  grid = (N/256, M/128, 3), block = 256 (8 waves)
// Wave(w): rows (w&1)*64 + tm*16 (tm=0..3), cols (w>>1)*64 + tn*16 (tn=0..3)
// ---------------------------------------------------------------------------
__global__ __launch_bounds__(256)
void qkv_proj_kernel(const float* __restrict__ X,
                     const float* __restrict__ Wq, const float* __restrict__ Bq,
                     const float* __restrict__ Wk, const float* __restrict__ Bk,
                     const float* __restrict__ Wv, const float* __restrict__ Bv,
                     _Float16* __restrict__ Qo, _Float16* __restrict__ Ko,
                     _Float16* __restrict__ Vo)
{
    __shared__ __attribute__((aligned(32))) _Float16 Xs[128][32]; // row-major (k contig)
    __shared__ __attribute__((aligned(32))) _Float16 Ws[256][32]; // N-major  (k contig)

    const float* W;  const float* Bias;  _Float16* Out;
    if (blockIdx.z == 0)      { W = Wq; Bias = Bq; Out = Qo; }
    else if (blockIdx.z == 1) { W = Wk; Bias = Bk; Out = Ko; }
    else                      { W = Wv; Bias = Bv; Out = Vo; }

    const int m0   = blockIdx.y * 128;
    const int n0   = blockIdx.x * 256;
    const int t    = threadIdx.x;
    const int wave = t >> 5;
    const int lane = t & 31;
    const int half = lane >> 4;
    const int ln   = lane & 15;
    const int wm   = (wave & 1) * 64;
    const int wn   = (wave >> 1) * 64;

    v8f acc[4][4] = {};

    for (int k0 = 0; k0 < HID; k0 += 32) {
        // ---- stage X tile (128 rows x 32 k), f32 -> f16, row-major ----
        {
            const int r = t >> 3;            // 0..31
            const int c = (t & 7) * 4;       // 0..28
#pragma unroll
            for (int i = 0; i < 4; ++i) {
                const int row = i * 32 + r;
                const float4 x4 = *(const float4*)&X[(size_t)(m0 + row) * HID + k0 + c];
                v4h h4;
                h4[0] = (_Float16)x4.x; h4[1] = (_Float16)x4.y;
                h4[2] = (_Float16)x4.z; h4[3] = (_Float16)x4.w;
                *(v4h*)&Xs[row][c] = h4;
            }
        }
        // ---- stage W tile (32 k x 256 n), transposed to N-major ----
        {
            const int kr = t >> 6;           // 0..3
            const int nc = (t & 63) * 4;     // 0..252
#pragma unroll
            for (int i = 0; i < 8; ++i) {
                const int kk = i * 4 + kr;
                const float4 w4 = *(const float4*)&W[(size_t)(k0 + kk) * HID + n0 + nc];
                Ws[nc + 0][kk] = (_Float16)w4.x;
                Ws[nc + 1][kk] = (_Float16)w4.y;
                Ws[nc + 2][kk] = (_Float16)w4.z;
                Ws[nc + 3][kk] = (_Float16)w4.w;
            }
        }
        if (k0 + 32 < HID) {   // warm GL2 for the next k-step
            __builtin_prefetch(&X[(size_t)(m0 + (t >> 3)) * HID + k0 + 32], 0, 0);
            __builtin_prefetch(&W[(size_t)(k0 + 32 + (t >> 6)) * HID + n0 + (t & 63) * 4], 0, 0);
        }
        __syncthreads();

        // B fragments: lane(h,n) holds W[k=8h+0..7 and 16+8h+0..7][n]
        v16h bf[4];
#pragma unroll
        for (int tn = 0; tn < 4; ++tn) {
            AFrag t2;
            t2.h[0] = *(const v8h*)&Ws[wn + tn * 16 + ln][8 * half];
            t2.h[1] = *(const v8h*)&Ws[wn + tn * 16 + ln][16 + 8 * half];
            bf[tn] = t2.v;
        }

#pragma unroll
        for (int tm = 0; tm < 4; ++tm) {
            AFrag af;   // lane m: k = 8h+0..7 and 16+8h+0..7
            af.h[0] = *(const v8h*)&Xs[wm + tm * 16 + ln][8 * half];
            af.h[1] = *(const v8h*)&Xs[wm + tm * 16 + ln][16 + 8 * half];
#pragma unroll
            for (int tn = 0; tn < 4; ++tn)
                acc[tm][tn] = wmma_f16(af.v, bf[tn], acc[tm][tn]);
        }
        __syncthreads();
    }

    // ---- epilogue: add bias, store f16 ----
#pragma unroll
    for (int tm = 0; tm < 4; ++tm) {
#pragma unroll
        for (int tn = 0; tn < 4; ++tn) {
            const int col = n0 + wn + tn * 16 + ln;
            const float b = Bias[col];
#pragma unroll
            for (int v = 0; v < 8; ++v) {
                const int row = m0 + wm + tm * 16 + 8 * half + v;
                Out[(size_t)row * HID + col] = (_Float16)(acc[tm][tn][v] + b);
            }
        }
    }
}

// ---------------------------------------------------------------------------
// Kernel 2: flash attention.
// grid = (S/128, HEADS, BATCH), block = 256 (8 waves, 16 query rows each).
// 64-key blocks: S = Q*K^T (4 n-tiles x 2 k-chunks = 8 WMMA),
// online softmax, ctx += P*V (4 n-tiles x 2 k-chunks = 8 WMMA).
// Q and K tiles staged with GLOBAL_LOAD_ASYNC_TO_LDS_B128 (ASYNCcnt).
// ---------------------------------------------------------------------------
__global__ __launch_bounds__(256)
void attention_kernel(const _Float16* __restrict__ Q,
                      const _Float16* __restrict__ K,
                      const _Float16* __restrict__ V,
                      float* __restrict__ Out)
{
    __shared__ __attribute__((aligned(32))) _Float16 Qs[128][64];    // [q][d]
    __shared__ __attribute__((aligned(32))) _Float16 Ks[64][64];     // [key][d] (B=K^T: d contig)
    __shared__ __attribute__((aligned(32))) _Float16 Vs[64][64];     // [d][key] (B=V : key contig)
    __shared__ __attribute__((aligned(32))) _Float16 Ps[8][16][64];  // per-wave P scratch

    const int qt = blockIdx.x;           // query tile (128 rows)
    const int h  = blockIdx.y;
    const int b  = blockIdx.z;

    const int t    = threadIdx.x;
    const int wave = t >> 5;
    const int lane = t & 31;
    const int half = lane >> 4;
    const int ln   = lane & 15;
    const int qw   = wave * 16;

    const size_t rowbase = (size_t)b * SEQ;            // row in [8192 x 1024] layout
    const _Float16* Qh = Q + (rowbase + qt * 128) * HID + h * HD;
    const _Float16* Kh = K + rowbase * HID + h * HD;
    const _Float16* Vh = V + rowbase * HID + h * HD;

    // ---- async stage Q tile (128 x 64 f16 = 16KB), 64B per thread ----
    {
        const int r  = t >> 1;           // 0..127
        const int d0 = (t & 1) * 32;     // 0 or 32 halves
        const unsigned lq = lds_off(&Qs[r][d0]);
        const unsigned long long gq = (unsigned long long)&Qh[(size_t)r * HID + d0];
        asm volatile("global_load_async_to_lds_b128 %0, %1, off\n\t"
                     "global_load_async_to_lds_b128 %0, %1, off offset:16\n\t"
                     "global_load_async_to_lds_b128 %0, %1, off offset:32\n\t"
                     "global_load_async_to_lds_b128 %0, %1, off offset:48"
                     :: "v"(lq), "v"(gq) : "memory");
    }
    asm volatile("s_wait_asynccnt 0x0" ::: "memory");
    __syncthreads();

    // Q fragments (16x64 = two 16x32 A-frags), held for the whole pass
    AFrag qf[2];
#pragma unroll
    for (int kc = 0; kc < 2; ++kc) {
        qf[kc].h[0] = *(const v8h*)&Qs[qw + ln][kc * 32 + 8 * half];
        qf[kc].h[1] = *(const v8h*)&Qs[qw + ln][kc * 32 + 16 + 8 * half];
    }

    v8f   ctx[4] = {};
    float m_r[8], l_r[8];
#pragma unroll
    for (int v = 0; v < 8; ++v) { m_r[v] = -3.402823466e38f; l_r[v] = 0.0f; }

    const float SCALE = 0.125f;          // 1/sqrt(64)

    for (int j0 = 0; j0 < SEQ; j0 += 64) {
        __syncthreads();                 // previous block fully consumed
        {
            const int key = t >> 2;          // 0..63
            const int d   = (t & 3) * 16;    // 0,16,32,48
            // ---- async K tile copy (natural [key][d] layout) ----
            const unsigned lk = lds_off(&Ks[key][d]);
            const unsigned long long gk =
                (unsigned long long)&Kh[(size_t)(j0 + key) * HID + d];
            asm volatile("global_load_async_to_lds_b128 %0, %1, off\n\t"
                         "global_load_async_to_lds_b128 %0, %1, off offset:16"
                         :: "v"(lk), "v"(gk) : "memory");
            // ---- V tile transposed to [d][key] (overlaps with async) ----
#pragma unroll
            for (int i2 = 0; i2 < 2; ++i2) {
                const v8h vv = *(const v8h*)&Vh[(size_t)(j0 + key) * HID + d + i2 * 8];
#pragma unroll
                for (int u = 0; u < 8; ++u) Vs[d + i2 * 8 + u][key] = vv[u];
            }
            if (j0 + 64 < SEQ) {             // warm GL2 for next key block
                __builtin_prefetch(&Kh[(size_t)(j0 + 64 + key) * HID + d], 0, 0);
                __builtin_prefetch(&Vh[(size_t)(j0 + 64 + key) * HID + d], 0, 0);
            }
        }
        asm volatile("s_wait_asynccnt 0x0" ::: "memory");
        __syncthreads();

        // ---- scores: S(16q x 64keys) = Q * K^T ----
        v8f s[4] = {};
#pragma unroll
        for (int nt = 0; nt < 4; ++nt) {
#pragma unroll
            for (int kc = 0; kc < 2; ++kc) {
                AFrag bf;
                bf.h[0] = *(const v8h*)&Ks[nt * 16 + ln][kc * 32 + 8 * half];
                bf.h[1] = *(const v8h*)&Ks[nt * 16 + ln][kc * 32 + 16 + 8 * half];
                s[nt] = wmma_f16(qf[kc].v, bf.v, s[nt]);
            }
        }

        // ---- online softmax (rows live in VGPR index v, cols in lanes) ----
#pragma unroll
        for (int v = 0; v < 8; ++v) {
            const float s0 = s[0][v] * SCALE;
            const float s1 = s[1][v] * SCALE;
            const float s2 = s[2][v] * SCALE;
            const float s3 = s[3][v] * SCALE;
            float mx = fmaxf(fmaxf(s0, s1), fmaxf(s2, s3));
#pragma unroll
            for (int off = 1; off < 16; off <<= 1)
                mx = fmaxf(mx, __shfl_xor(mx, off, 32));
            const float mn   = fmaxf(m_r[v], mx);
            const float corr = __expf(m_r[v] - mn);
            const float p0 = __expf(s0 - mn);
            const float p1 = __expf(s1 - mn);
            const float p2 = __expf(s2 - mn);
            const float p3 = __expf(s3 - mn);
            float rs = (p0 + p1) + (p2 + p3);
#pragma unroll
            for (int off = 1; off < 16; off <<= 1)
                rs += __shfl_xor(rs, off, 32);
            l_r[v] = l_r[v] * corr + rs;
            m_r[v] = mn;
#pragma unroll
            for (int j = 0; j < 4; ++j) ctx[j][v] *= corr;
            // C-layout -> LDS (row = v + 8*half, col = key)
            Ps[wave][8 * half + v][ln]      = (_Float16)p0;
            Ps[wave][8 * half + v][16 + ln] = (_Float16)p1;
            Ps[wave][8 * half + v][32 + ln] = (_Float16)p2;
            Ps[wave][8 * half + v][48 + ln] = (_Float16)p3;
        }

        // ---- P as A-fragments (16x64 = two 16x32), ctx += P * V ----
        AFrag pf[2];
#pragma unroll
        for (int kc = 0; kc < 2; ++kc) {
            pf[kc].h[0] = *(const v8h*)&Ps[wave][ln][kc * 32 + 8 * half];
            pf[kc].h[1] = *(const v8h*)&Ps[wave][ln][kc * 32 + 16 + 8 * half];
        }
#pragma unroll
        for (int j = 0; j < 4; ++j) {
#pragma unroll
            for (int kc = 0; kc < 2; ++kc) {
                AFrag vf;
                vf.h[0] = *(const v8h*)&Vs[j * 16 + ln][kc * 32 + 8 * half];
                vf.h[1] = *(const v8h*)&Vs[j * 16 + ln][kc * 32 + 16 + 8 * half];
                ctx[j] = wmma_f16(pf[kc].v, vf.v, ctx[j]);
            }
        }
    }

    // ---- epilogue: normalize and store f32 ----
#pragma unroll
    for (int v = 0; v < 8; ++v) {
        const float invl = 1.0f / l_r[v];
        const int   q    = qt * 128 + qw + 8 * half + v;
        float* orow = Out + ((size_t)b * SEQ + q) * HID + h * HD;
#pragma unroll
        for (int j = 0; j < 4; ++j)
            orow[j * 16 + ln] = ctx[j][v] * invl;
    }
}

// ---------------------------------------------------------------------------
extern "C" void kernel_launch(void* const* d_in, const int* in_sizes, int n_in,
                              void* d_out, int out_size, void* d_ws, size_t ws_size,
                              hipStream_t stream) {
    const float* X  = (const float*)d_in[0];
    const float* Wq = (const float*)d_in[1];
    const float* Bq = (const float*)d_in[2];
    const float* Wk = (const float*)d_in[3];
    const float* Bk = (const float*)d_in[4];
    const float* Wv = (const float*)d_in[5];
    const float* Bv = (const float*)d_in[6];

    _Float16* Qw = (_Float16*)d_ws;                       // 16 MB
    _Float16* Kw = Qw + (size_t)MROWS * HID;              // 16 MB
    _Float16* Vw = Kw + (size_t)MROWS * HID;              // 16 MB

    dim3 g1(HID / 256, MROWS / 128, 3), b1(256);
    qkv_proj_kernel<<<g1, b1, 0, stream>>>(X, Wq, Bq, Wk, Bk, Wv, Bv, Qw, Kw, Vw);

    dim3 g2(SEQ / 128, HEADS, BATCH), b2(256);
    attention_kernel<<<g2, b2, 0, stream>>>(Qw, Kw, Vw, (float*)d_out);
}
